// UpsampleAttention_51943334477849
// MI455X (gfx1250) — matrix operand order repaired and checked by reading, not compiled, not run on hardware
//
#include <hip/hip_runtime.h>
#include <hip/hip_bf16.h>

typedef unsigned short u16;
typedef __attribute__((ext_vector_type(16))) __bf16 v16bf;
typedef __attribute__((ext_vector_type(8)))  float  v8f;

#define HEADS 8
#define UPF   2

// ---------- helpers ----------

__device__ __forceinline__ u16 f2bf(float f) {
  unsigned int u = __float_as_uint(f);
  u += 0x7FFFu + ((u >> 16) & 1u);   // round-to-nearest-even
  return (u16)(u >> 16);
}

union FragU { uint4 u4[2]; v16bf bf; };

// A fragment (16x32 bf16): lanes<16 hold K {0..15}, lanes>=16 hold K {16..31} (contiguous)
__device__ __forceinline__ v16bf frag_a(const u16* p, int hi) {
  FragU f;
  f.u4[0] = *reinterpret_cast<const uint4*>(p + 16 * hi);
  f.u4[1] = *reinterpret_cast<const uint4*>(p + 16 * hi + 8);
  return f.bf;
}
// B fragment (32x16 bf16): lane holds col (lane&15), K = 16*hi + i (contiguous)
__device__ __forceinline__ v16bf frag_b(const u16* p) {
  FragU f;
  f.u4[0] = *reinterpret_cast<const uint4*>(p);
  f.u4[1] = *reinterpret_cast<const uint4*>(p + 8);
  return f.bf;
}
__device__ __forceinline__ v8f wmma_bf16(v16bf a, v16bf b, v8f c) {
  return __builtin_amdgcn_wmma_f32_16x16x32_bf16(false, a, false, b, (short)0, c,
                                                 false, false);
}

// ---------- elementwise prep ----------

__global__ void cvt_bf16(const float* __restrict__ src, u16* __restrict__ dst, int n) {
  int i = blockIdx.x * blockDim.x + threadIdx.x;
  if (i < n) dst[i] = f2bf(src[i]);
}

// linear interpolation upsample (align_corners=False), x:[B,L,D] -> xu:[B,2L,D] (bf16)
__global__ void upsample_bf16(const float* __restrict__ x, u16* __restrict__ xu,
                              int Bn, int Ln, int Dm) {
  int idx = blockIdx.x * blockDim.x + threadIdx.x;
  int total = Bn * Ln * UPF * Dm;
  if (idx >= total) return;
  int d = idx % Dm;
  int t = idx / Dm;
  int q = t % (Ln * UPF);
  int b = t / (Ln * UPF);
  float src = fmaxf((q + 0.5f) * (1.0f / UPF) - 0.5f, 0.0f);
  int i0 = (int)src;
  int i1 = min(i0 + 1, Ln - 1);
  float w  = src - (float)i0;
  float v = x[(size_t)(b * Ln + i0) * Dm + d] * (1.0f - w) +
            x[(size_t)(b * Ln + i1) * Dm + d] * w;
  xu[idx] = f2bf(v);
}

// ---------- tiled WMMA GEMM: C[M,N] = A[M,K] * W[N,K]^T + bias ----------
// 128x128 block, 8 waves (2x4), 64x32 per wave, K-step 32, double-buffered LDS.

#define GT_K 32
#define ASTR 40  // 32 + 8 halves pad; row = 80B, 16B aligned

template <bool OUT_BF16>
__global__ __launch_bounds__(256)
void gemm_bf16_nt(const u16* __restrict__ A, const u16* __restrict__ W,
                  const float* __restrict__ bias, void* __restrict__ Cout,
                  int M, int N, int K) {
  __shared__ __align__(16) u16 sA[2][128 * ASTR];
  __shared__ __align__(16) u16 sB[2][128 * ASTR];
  const int tid   = threadIdx.x;
  const int lane  = tid & 31;
  const int wvid  = tid >> 5;
  const int waveM = wvid >> 2;   // 0..1 : 64 rows each
  const int waveN = wvid & 3;    // 0..3 : 32 cols each
  const int hi    = lane >> 4;
  const int l15   = lane & 15;
  const int mBlk  = blockIdx.y * 128;
  const int nBlk  = blockIdx.x * 128;
  const int ldRow = tid >> 1;          // 0..127
  const int ldCol = (tid & 1) * 16;    // 0 or 16 halves

  const u16* gArow = A + (size_t)(mBlk + ldRow) * K + ldCol;
  const u16* gWrow = W + (size_t)(nBlk + ldRow) * K + ldCol;
  const int ldsOff = ldRow * ASTR + ldCol;

  const v8f vzero = {};
  v8f acc[4][2];
#pragma unroll
  for (int i = 0; i < 4; ++i)
#pragma unroll
    for (int j = 0; j < 2; ++j) acc[i][j] = vzero;

  // stage k-tile 0 into buffer 0
  {
    uint4 a0 = *reinterpret_cast<const uint4*>(gArow);
    uint4 a1 = *reinterpret_cast<const uint4*>(gArow + 8);
    uint4 b0 = *reinterpret_cast<const uint4*>(gWrow);
    uint4 b1 = *reinterpret_cast<const uint4*>(gWrow + 8);
    *reinterpret_cast<uint4*>(&sA[0][ldsOff])     = a0;
    *reinterpret_cast<uint4*>(&sA[0][ldsOff + 8]) = a1;
    *reinterpret_cast<uint4*>(&sB[0][ldsOff])     = b0;
    *reinterpret_cast<uint4*>(&sB[0][ldsOff + 8]) = b1;
  }
  __syncthreads();

  int cur = 0;
  for (int k0 = 0; k0 < K; k0 += GT_K) {
    const bool more = (k0 + GT_K) < K;
    uint4 a0, a1, b0, b1;
    if (more) {  // issue next tile's global loads before compute
      a0 = *reinterpret_cast<const uint4*>(gArow + k0 + GT_K);
      a1 = *reinterpret_cast<const uint4*>(gArow + k0 + GT_K + 8);
      b0 = *reinterpret_cast<const uint4*>(gWrow + k0 + GT_K);
      b1 = *reinterpret_cast<const uint4*>(gWrow + k0 + GT_K + 8);
    }

    // gather all fragments, then back-to-back WMMAs
    v16bf bf[2], af[4];
    bf[0] = frag_b(&sB[cur][(waveN * 32 +  0 + l15) * ASTR + 16 * hi]);
    bf[1] = frag_b(&sB[cur][(waveN * 32 + 16 + l15) * ASTR + 16 * hi]);
#pragma unroll
    for (int mt = 0; mt < 4; ++mt)
      af[mt] = frag_a(&sA[cur][(waveM * 64 + mt * 16 + l15) * ASTR], hi);
#pragma unroll
    for (int mt = 0; mt < 4; ++mt) {
      acc[mt][0] = wmma_bf16(af[mt], bf[0], acc[mt][0]);
      acc[mt][1] = wmma_bf16(af[mt], bf[1], acc[mt][1]);
    }

    if (more) {  // store next tile to the other buffer
      *reinterpret_cast<uint4*>(&sA[cur ^ 1][ldsOff])     = a0;
      *reinterpret_cast<uint4*>(&sA[cur ^ 1][ldsOff + 8]) = a1;
      *reinterpret_cast<uint4*>(&sB[cur ^ 1][ldsOff])     = b0;
      *reinterpret_cast<uint4*>(&sB[cur ^ 1][ldsOff + 8]) = b1;
    }
    __syncthreads();
    cur ^= 1;
  }

#pragma unroll
  for (int nt = 0; nt < 2; ++nt) {
    const int col = nBlk + waveN * 32 + nt * 16 + l15;
    const float bv = bias[col];
#pragma unroll
    for (int mt = 0; mt < 4; ++mt) {
#pragma unroll
      for (int i = 0; i < 8; ++i) {
        const int row = mBlk + waveM * 64 + mt * 16 + i + 8 * hi;
        float v = acc[mt][nt][i] + bv;
        if (OUT_BF16)
          ((u16*)Cout)[(size_t)row * N + col] = f2bf(v);
        else
          ((float*)Cout)[(size_t)row * N + col] = v;
      }
    }
  }
}

// ---------- flash attention: per (b,h,64-row q tile), stream 64-wide K/V tiles ----------

#define KSTR 72  // 64 + 8 halves pad; row = 144B, 16B aligned

union VTU { uint4 u4[4]; u16 s[32]; };

__global__ __launch_bounds__(128)
void attn_flash(const u16* __restrict__ qg, const u16* __restrict__ kg,
                const u16* __restrict__ vg, u16* __restrict__ og,
                int Bn, int Qn, int Ln, int Dm) {
  __shared__ __align__(16) u16 sQ[64 * KSTR];
  __shared__ __align__(16) u16 sK[2][64 * KSTR];
  __shared__ __align__(16) u16 sVt[2][64 * KSTR];   // transposed: [d][kpos]
  __shared__ __align__(16) u16 sP[4][16 * KSTR];    // per-wave P staging

  const int tid  = threadIdx.x;
  const int lane = tid & 31;
  const int wvid = tid >> 5;     // 4 waves, one 16-row q tile each
  const int hi   = lane >> 4;
  const int l15  = lane & 15;
  const int qB   = blockIdx.x * 64;
  const int h    = blockIdx.y;
  const int b    = blockIdx.z;
  const int hOff = h * 64;
  const int ldRow = tid >> 1;        // 0..63
  const int ldCol = (tid & 1) * 32;  // 0 or 32 halves
  const int ldsOff = ldRow * KSTR + ldCol;

  const u16* gKbase = kg + (size_t)b * Ln * Dm + hOff + ldCol;
  const u16* gVbase = vg + (size_t)b * Ln * Dm + hOff + ldCol;

  // stage Q tile (64 x 64 halves)
  {
    const uint4* g = reinterpret_cast<const uint4*>(
        qg + (size_t)(b * Qn + qB + ldRow) * Dm + hOff + ldCol);
    uint4 q0 = g[0], q1 = g[1], q2 = g[2], q3 = g[3];
    uint4* d = reinterpret_cast<uint4*>(&sQ[ldsOff]);
    d[0] = q0; d[1] = q1; d[2] = q2; d[3] = q3;
  }
  // stage K/V block 0 into buffer 0
  {
    const uint4* gk = reinterpret_cast<const uint4*>(gKbase + (size_t)ldRow * Dm);
    uint4 k0 = gk[0], k1 = gk[1], k2 = gk[2], k3 = gk[3];
    VTU vt;
    const uint4* gv = reinterpret_cast<const uint4*>(gVbase + (size_t)ldRow * Dm);
    vt.u4[0] = gv[0]; vt.u4[1] = gv[1]; vt.u4[2] = gv[2]; vt.u4[3] = gv[3];
    uint4* dk = reinterpret_cast<uint4*>(&sK[0][ldsOff]);
    dk[0] = k0; dk[1] = k1; dk[2] = k2; dk[3] = k3;
#pragma unroll
    for (int j = 0; j < 32; ++j)
      sVt[0][(ldCol + j) * KSTR + ldRow] = vt.s[j];
  }
  __syncthreads();

  // Q fragments are loop-invariant: hoist to registers
  const v16bf qa0 = frag_a(&sQ[(wvid * 16 + l15) * KSTR +  0], hi);
  const v16bf qa1 = frag_a(&sQ[(wvid * 16 + l15) * KSTR + 32], hi);

  const v8f vzero = {};
  v8f o[4];
  float mrow[8], lrow[8];
#pragma unroll
  for (int i = 0; i < 4; ++i) o[i] = vzero;
#pragma unroll
  for (int i = 0; i < 8; ++i) { mrow[i] = -3.0e38f; lrow[i] = 0.0f; }

  const int qW = qB + wvid * 16;
  const int kEnd = min(Ln, qB + 64);           // causal: k <= q
  const float scale = 0.044194173824159216f;   // 512^-0.5 (full-D scaling per ref)

  int cur = 0;
  for (int kB = 0; kB < kEnd; kB += 64) {
    const bool more = (kB + 64) < kEnd;
    uint4 kr0, kr1, kr2, kr3;
    VTU vt;
    if (more) {  // issue next block's global loads before compute
      const uint4* gk = reinterpret_cast<const uint4*>(
          gKbase + (size_t)(kB + 64 + ldRow) * Dm);
      kr0 = gk[0]; kr1 = gk[1]; kr2 = gk[2]; kr3 = gk[3];
      const uint4* gv = reinterpret_cast<const uint4*>(
          gVbase + (size_t)(kB + 64 + ldRow) * Dm);
      vt.u4[0] = gv[0]; vt.u4[1] = gv[1]; vt.u4[2] = gv[2]; vt.u4[3] = gv[3];
    }

    // S = Q * K^T  (hd=64 reduction -> 2 WMMA k-steps, 4 kpos subtiles)
    v8f s[4];
    s[0] = vzero; s[1] = vzero; s[2] = vzero; s[3] = vzero;
    {
      v16bf kb[4];
#pragma unroll
      for (int t = 0; t < 4; ++t)
        kb[t] = frag_b(&sK[cur][(t * 16 + l15) * KSTR + 16 * hi]);
#pragma unroll
      for (int t = 0; t < 4; ++t) s[t] = wmma_bf16(qa0, kb[t], s[t]);
#pragma unroll
      for (int t = 0; t < 4; ++t)
        kb[t] = frag_b(&sK[cur][(t * 16 + l15) * KSTR + 32 + 16 * hi]);
#pragma unroll
      for (int t = 0; t < 4; ++t) s[t] = wmma_bf16(qa1, kb[t], s[t]);
    }

    // scale + causal mask (element: kpos > qrow)
    const bool needMask = (kB + 63) > qW;
#pragma unroll
    for (int t = 0; t < 4; ++t) {
#pragma unroll
      for (int i = 0; i < 8; ++i) {
        float v = s[t][i] * scale;
        if (needMask) {
          int kp = kB + t * 16 + l15;
          int qr = qW + i + 8 * hi;
          if (kp > qr) v = -1.0e30f;
        }
        s[t][i] = v;
      }
    }

    // online softmax: rows live in one 16-lane half -> shuffle-xor 1,2,4,8
#pragma unroll
    for (int i = 0; i < 8; ++i) {
      float v = fmaxf(fmaxf(s[0][i], s[1][i]), fmaxf(s[2][i], s[3][i]));
      v = fmaxf(v, __shfl_xor(v, 1, 32));
      v = fmaxf(v, __shfl_xor(v, 2, 32));
      v = fmaxf(v, __shfl_xor(v, 4, 32));
      v = fmaxf(v, __shfl_xor(v, 8, 32));
      float mn = fmaxf(mrow[i], v);
      float al = __expf(mrow[i] - mn);
      mrow[i] = mn;
      lrow[i] *= al;
      o[0][i] *= al; o[1][i] *= al; o[2][i] *= al; o[3][i] *= al;
      float r = 0.0f;
#pragma unroll
      for (int t = 0; t < 4; ++t) {
        float p = __expf(s[t][i] - mn);
        s[t][i] = p;
        r += p;
      }
      r += __shfl_xor(r, 1, 32);
      r += __shfl_xor(r, 2, 32);
      r += __shfl_xor(r, 4, 32);
      r += __shfl_xor(r, 8, 32);
      lrow[i] += r;
    }

    // C-layout -> A-layout via per-wave LDS (same-wave DS ops stay in order)
#pragma unroll
    for (int t = 0; t < 4; ++t)
#pragma unroll
      for (int i = 0; i < 8; ++i)
        sP[wvid][(i + 8 * hi) * KSTR + t * 16 + l15] = f2bf(s[t][i]);

    // O += P * V  (kpos=64 reduction -> 2 WMMA k-steps, 4 hd subtiles)
    {
      v16bf vb[4];
      const v16bf pa0 = frag_a(&sP[wvid][l15 * KSTR], hi);
#pragma unroll
      for (int dt = 0; dt < 4; ++dt)
        vb[dt] = frag_b(&sVt[cur][(dt * 16 + l15) * KSTR + 16 * hi]);
#pragma unroll
      for (int dt = 0; dt < 4; ++dt) o[dt] = wmma_bf16(pa0, vb[dt], o[dt]);
      const v16bf pa1 = frag_a(&sP[wvid][l15 * KSTR + 32], hi);
#pragma unroll
      for (int dt = 0; dt < 4; ++dt)
        vb[dt] = frag_b(&sVt[cur][(dt * 16 + l15) * KSTR + 32 + 16 * hi]);
#pragma unroll
      for (int dt = 0; dt < 4; ++dt) o[dt] = wmma_bf16(pa1, vb[dt], o[dt]);
    }

    if (more) {  // store next K/V block to the other buffer
      uint4* dk = reinterpret_cast<uint4*>(&sK[cur ^ 1][ldsOff]);
      dk[0] = kr0; dk[1] = kr1; dk[2] = kr2; dk[3] = kr3;
#pragma unroll
      for (int j = 0; j < 32; ++j)
        sVt[cur ^ 1][(ldCol + j) * KSTR + ldRow] = vt.s[j];
    }
    __syncthreads();
    cur ^= 1;
  }

  // epilogue: O /= l, write bf16 att[b, q, h*64 + d]
#pragma unroll
  for (int dt = 0; dt < 4; ++dt)
#pragma unroll
    for (int i = 0; i < 8; ++i) {
      int qr = qW + i + 8 * hi;
      float v = o[dt][i] / lrow[i];
      og[(size_t)(b * Qn + qr) * Dm + hOff + dt * 16 + l15] = f2bf(v);
    }
}

// ---------- launch ----------

extern "C" void kernel_launch(void* const* d_in, const int* in_sizes, int n_in,
                              void* d_out, int out_size, void* d_ws, size_t ws_size,
                              hipStream_t stream) {
  const int B = 2, L = 2048, D = 512, Q = L * UPF;

  const float* x  = (const float*)d_in[0];
  const float* Wq = (const float*)d_in[1];
  const float* bq = (const float*)d_in[2];
  const float* Wk = (const float*)d_in[3];
  const float* bk = (const float*)d_in[4];
  const float* Wv = (const float*)d_in[5];
  const float* bv = (const float*)d_in[6];
  const float* Wo = (const float*)d_in[7];
  const float* bo = (const float*)d_in[8];

  u16* p = (u16*)d_ws;
  u16* x_bf  = p; p += (size_t)B * L * D;
  u16* xu_bf = p; p += (size_t)B * Q * D;
  u16* Wq_bf = p; p += (size_t)D * D;
  u16* Wk_bf = p; p += (size_t)D * D;
  u16* Wv_bf = p; p += (size_t)D * D;
  u16* Wo_bf = p; p += (size_t)D * D;
  u16* q_bf  = p; p += (size_t)B * Q * D;
  u16* k_bf  = p; p += (size_t)B * L * D;
  u16* v_bf  = p; p += (size_t)B * L * D;
  u16* a_bf  = p; p += (size_t)B * Q * D;

  const int nx = B * L * D;
  const int nw = D * D;
  const int nu = B * Q * D;
  cvt_bf16<<<(nx + 255) / 256, 256, 0, stream>>>(x,  x_bf,  nx);
  cvt_bf16<<<(nw + 255) / 256, 256, 0, stream>>>(Wq, Wq_bf, nw);
  cvt_bf16<<<(nw + 255) / 256, 256, 0, stream>>>(Wk, Wk_bf, nw);
  cvt_bf16<<<(nw + 255) / 256, 256, 0, stream>>>(Wv, Wv_bf, nw);
  cvt_bf16<<<(nw + 255) / 256, 256, 0, stream>>>(Wo, Wo_bf, nw);
  upsample_bf16<<<(nu + 255) / 256, 256, 0, stream>>>(x, xu_bf, B, L, D);

  gemm_bf16_nt<true><<<dim3(D / 128, (B * Q) / 128), dim3(256), 0, stream>>>(
      xu_bf, Wq_bf, bq, q_bf, B * Q, D, D);
  gemm_bf16_nt<true><<<dim3(D / 128, (B * L) / 128), dim3(256), 0, stream>>>(
      x_bf, Wk_bf, bk, k_bf, B * L, D, D);
  gemm_bf16_nt<true><<<dim3(D / 128, (B * L) / 128), dim3(256), 0, stream>>>(
      x_bf, Wv_bf, bv, v_bf, B * L, D, D);

  attn_flash<<<dim3(Q / 64, HEADS, B), dim3(128), 0, stream>>>(
      q_bf, k_bf, v_bf, a_bf, B, Q, L, D);

  gemm_bf16_nt<false><<<dim3(D / 128, (B * Q) / 128), dim3(256), 0, stream>>>(
      a_bf, Wo_bf, bo, d_out, B * Q, D, D);
}
